// JointNetwork_85126251806848
// MI455X (gfx1250) — matrix operand order repaired and checked by reading, not compiled, not run on hardware
//
#include <hip/hip_runtime.h>
#include <hip/hip_bf16.h>

typedef __attribute__((ext_vector_type(16))) __bf16 v16bf;
typedef __attribute__((ext_vector_type(8)))  __bf16 v8bf;
typedef __attribute__((ext_vector_type(8)))  float  v8f;

#define TB 200
#define UB 50
#define DIN 512
#define JD 640
#define VOC 600
#define NTP 40     // N tiles padded to 40 so each of 8 waves owns exactly 5
#define NPW 5      // N tiles per wave
#define KC 20      // 640/32 K chunks
#define ZSTRIDE 648
#define MB 64      // M rows per block (4 WMMA M-tiles)
#define MTILES 4

// ---------------- Kernel 1: row-parallel projection (E = X*W + b) ----------------
__global__ void proj640(const float* __restrict__ X, const float* __restrict__ W,
                        const float* __restrict__ bias, float* __restrict__ out, int Din) {
  const int j = threadIdx.x;           // 0..639
  const int row = blockIdx.x;
  const float* x = X + (long)row * Din;
  float acc = bias ? bias[j] : 0.0f;
#pragma unroll 8
  for (int d = 0; d < Din; ++d)
    acc = fmaf(x[d], W[d * JD + j], acc);
  out[(long)row * JD + j] = acc;
}

// ---------------- Kernel 2: pack W_out into WMMA B-fragment layout, hi/lo bf16 ----------------
// B in contiguous k order (as A): lane<16 -> N=lane, K=kchunk*32+0..15 ; lane>=16 -> N=lane-16, K=+16..31
// Padded to NTP=40 N-tiles; columns >= 600 are zero.
__global__ void packW(const float* __restrict__ W, __bf16* __restrict__ whi,
                      __bf16* __restrict__ wlo) {
  const int group = blockIdx.x * 8 + (threadIdx.x >> 5);   // 0..799 = (ntile,kchunk)
  const int lane  = threadIdx.x & 31;
  const int ntile = group / KC;
  const int kchunk = group % KC;
  const int n = ntile * 16 + (lane & 15);
  const int kbase = kchunk * 32 + ((lane < 16) ? 0 : 16);
  const long base = ((long)group * 32 + lane) * 16;        // 16 bf16 per lane
#pragma unroll
  for (int i = 0; i < 16; ++i) {
    const int k = kbase + i;
    const float w = (n < VOC) ? W[k * VOC + n] : 0.0f;
    const __bf16 h = (__bf16)w;
    const __bf16 l = (__bf16)(w - (float)h);
    whi[base + i] = h;
    wlo[base + i] = l;
  }
}

// ---------------- Kernel 3: main joint GEMM via split-bf16 WMMA ----------------
// block = 256 threads (8 wave32). Block owns 64 M-rows (4 WMMA M-tiles).
// Loop nest: kc OUTER (A-fragments loop-local), nt x mt fully unrolled inside.
// Accumulators acc[5][4] live across kc; nothing nt-invariant remains hoistable.
__global__ void __launch_bounds__(256)
joint_main(const float* __restrict__ E, const float* __restrict__ D,
           const __bf16* __restrict__ whi, const __bf16* __restrict__ wlo,
           const float* __restrict__ b_out, float* __restrict__ out) {
  __shared__ __align__(16) __bf16 zhi[MB * ZSTRIDE];   // 82,944 B
  __shared__ __align__(16) __bf16 zlo[MB * ZSTRIDE];   // 82,944 B

  const int tid = threadIdx.x;
  const int mbase = blockIdx.x * MB;

  // ---- Phase 1: Z = tanh(E[row] + D[row]) for MB rows, split hi/lo bf16 into LDS ----
  for (int r = 0; r < MB; ++r) {
    const int m = mbase + r;
    const int erow = m / UB;                 // b*T + t
    const int u = m - erow * UB;
    const int b = erow / TB;
    const float* ep = E + (long)erow * JD;
    const float* dp = D + (long)(b * UB + u) * JD;
    for (int k = tid; k < JD; k += 256) {
      const float z = tanhf(ep[k] + dp[k]);
      const __bf16 h = (__bf16)z;
      const __bf16 l = (__bf16)(z - (float)h);
      zhi[r * ZSTRIDE + k] = h;
      zlo[r * ZSTRIDE + k] = l;
    }
  }
  __syncthreads();

  // ---- Phase 2: kc-outer WMMA loop ----
  const int wave = tid >> 5;
  const int lane = tid & 31;
  const int mrow = lane & 15;                 // A-matrix: M = lane%16
  const int koff = (lane < 16) ? 0 : 16;      // lanes 0-15: K 0-15, lanes 16-31: K 16-31

  v8f acc[NPW][MTILES];
#pragma unroll
  for (int i = 0; i < NPW; ++i)
#pragma unroll
    for (int mt = 0; mt < MTILES; ++mt) acc[i][mt] = (v8f){};

  const v16bf* bhB = ((const v16bf*)whi) + lane;
  const v16bf* blB = ((const v16bf*)wlo) + lane;

  for (int kc = 0; kc < KC; ++kc) {
    // A fragments for this kc: 4 M-tiles, hi+lo (loop-local -> no cross-kc hoisting)
    union AU { v16bf v; v8bf h[2]; };
    AU ah[MTILES], al[MTILES];
#pragma unroll
    for (int mt = 0; mt < MTILES; ++mt) {
      const __bf16* zh0 = &zhi[(mt * 16 + mrow) * ZSTRIDE + koff + kc * 32];
      const __bf16* zl0 = &zlo[(mt * 16 + mrow) * ZSTRIDE + koff + kc * 32];
      ah[mt].h[0] = *(const v8bf*)(zh0);        // ds_read_b128
      ah[mt].h[1] = *(const v8bf*)(zh0 + 8);
      al[mt].h[0] = *(const v8bf*)(zl0);
      al[mt].h[1] = *(const v8bf*)(zl0 + 8);
    }
#pragma unroll
    for (int i = 0; i < NPW; ++i) {
      const int nt = wave + 8 * i;            // < 40
      const long bidx = ((long)nt * KC + kc) * 32;
      const v16bf bhv = bhB[bidx];            // 2x global_load_b128 (L2-resident)
      const v16bf blv = blB[bidx];
#pragma unroll
      for (int mt = 0; mt < MTILES; ++mt) {
        // fp32-accurate product: Ahi*Bhi + Ahi*Blo + Alo*Bhi
        acc[i][mt] = __builtin_amdgcn_wmma_f32_16x16x32_bf16(false, ah[mt].v, false, bhv, (short)0, acc[i][mt], false, false);
        acc[i][mt] = __builtin_amdgcn_wmma_f32_16x16x32_bf16(false, ah[mt].v, false, blv, (short)0, acc[i][mt], false, false);
        acc[i][mt] = __builtin_amdgcn_wmma_f32_16x16x32_bf16(false, al[mt].v, false, bhv, (short)0, acc[i][mt], false, false);
      }
    }
  }

  // ---- store: C/D layout: VGPR r -> M = r + 8*(lane>=16), N = lane%16 ----
#pragma unroll
  for (int i = 0; i < NPW; ++i) {
    const int nt = wave + 8 * i;
    const int n = nt * 16 + (lane & 15);
    if (n < VOC) {
      const float bo = b_out[n];
#pragma unroll
      for (int mt = 0; mt < MTILES; ++mt) {
        const int m0 = mbase + mt * 16 + ((lane < 16) ? 0 : 8);
#pragma unroll
        for (int r = 0; r < 8; ++r)
          out[(long)(m0 + r) * VOC + n] = acc[i][mt][r] + bo;
      }
    }
  }
}

extern "C" void kernel_launch(void* const* d_in, const int* in_sizes, int n_in,
                              void* d_out, int out_size, void* d_ws, size_t ws_size,
                              hipStream_t stream) {
  const float* h_enc = (const float*)d_in[0];   // (4,200,1,512)
  const float* h_dec = (const float*)d_in[1];   // (4,1,50,512)
  const float* W_enc = (const float*)d_in[2];   // (512,640)
  const float* b_enc = (const float*)d_in[3];   // (640)
  const float* W_dec = (const float*)d_in[4];   // (512,640)
  const float* W_out = (const float*)d_in[5];   // (640,600)
  const float* b_out = (const float*)d_in[6];   // (600)
  float* out = (float*)d_out;                   // (4,200,50,600)

  char* ws = (char*)d_ws;
  float*  Ebuf = (float*)(ws);                  // 800*640 f32    = 2,048,000 B
  float*  Dbuf = (float*)(ws + 2048000);        // 200*640 f32    =   512,000 B
  __bf16* whi  = (__bf16*)(ws + 2560000);       // 40*20*512 bf16 =   819,200 B
  __bf16* wlo  = (__bf16*)(ws + 3379200);       //                    819,200 B

  proj640<<<800, 640, 0, stream>>>(h_enc, W_enc, b_enc, Ebuf, DIN);
  proj640<<<200, 640, 0, stream>>>(h_dec, W_dec, nullptr, Dbuf, DIN);
  packW<<<100, 256, 0, stream>>>(W_out, whi, wlo);
  joint_main<<<625, 256, 0, stream>>>(Ebuf, Dbuf, whi, wlo, b_out, out);
}
